// TinyMyo8Ch400Quant_79182017069454
// MI455X (gfx1250) — 16-bit and 8-bit matrix operand order repaired and checked by reading, compiled, not run on hardware
//
#include <hip/hip_runtime.h>
#include <math.h>

constexpr int B_    = 256;
constexpr int C_    = 8;
constexpr int T_    = 1000;
constexpr int PATCH_= 20;
constexpr int D_    = 192;
constexpr int SEQ_  = 400;
constexpr int L_    = 8;
constexpr int H_    = 3;
constexpr int DH_   = 64;
constexpr int HID_  = 768;
constexpr int NCLS_ = 7;
constexpr int M_    = B_ * SEQ_;
constexpr int SPAD_ = 448;

typedef __attribute__((ext_vector_type(16))) _Float16 v16h;
typedef __attribute__((ext_vector_type(8)))  float    v8f;
typedef __attribute__((ext_vector_type(8)))  int      v8i;

__device__ __forceinline__ float fq_val(float x, float amax) {
  float s = amax * (1.f / 127.f) + 1e-8f;
  float q = fminf(fmaxf(rintf(x / s), -128.f), 127.f);
  return q * s;
}
__device__ __forceinline__ signed char q_i8(float x, float s) {
  return (signed char)fminf(fmaxf(rintf(x / s), -128.f), 127.f);
}
__device__ __forceinline__ float gelu_f(float x) {
  return 0.5f * x * (1.f + erff(x * 0.70710678118f));
}
__device__ __forceinline__ void async_cp16(unsigned lds_off, const void* gsrc) {
  unsigned long long src = (unsigned long long)(size_t)gsrc;
  asm volatile("global_load_async_to_lds_b128 %0, %1, off" :: "v"(lds_off), "v"(src) : "memory");
}
__device__ __forceinline__ void async_wait0() {
  asm volatile("s_wait_asynccnt 0x0" ::: "memory");
}

__global__ void k_zeroN(float* p, int n) {
  int i = blockIdx.x * blockDim.x + threadIdx.x;
  if (i < n) p[i] = 0.f;
}

__global__ __launch_bounds__(256)
void k_absmax(const float* __restrict__ x, long long n, float* __restrict__ slot) {
  float m = 0.f;
  for (long long i = blockIdx.x * 256LL + threadIdx.x; i < n; i += (long long)gridDim.x * 256)
    m = fmaxf(m, fabsf(x[i]));
  for (int off = 16; off; off >>= 1) m = fmaxf(m, __shfl_xor(m, off));
  if ((threadIdx.x & 31) == 0)
    atomicMax((unsigned int*)slot, __float_as_uint(m));
}

__global__ __launch_bounds__(256)
void k_absmax3(const float* __restrict__ qkv, long long rows, float* __restrict__ slots) {
  float m[3] = {0.f, 0.f, 0.f};
  long long n = rows * 576;
  for (long long i = blockIdx.x * 256LL + threadIdx.x; i < n; i += (long long)gridDim.x * 256) {
    int col = (int)(i % 576);
    int s = col / 192;
    m[s] = fmaxf(m[s], fabsf(qkv[i]));
  }
  for (int off = 16; off; off >>= 1)
    for (int j = 0; j < 3; ++j) m[j] = fmaxf(m[j], __shfl_xor(m[j], off));
  if ((threadIdx.x & 31) == 0)
    for (int j = 0; j < 3; ++j)
      atomicMax((unsigned int*)(slots + j), __float_as_uint(m[j]));
}

__global__ void k_fq16(const float* __restrict__ x, const float* __restrict__ amax,
                       _Float16* __restrict__ y, long long n) {
  long long i = blockIdx.x * 256LL + threadIdx.x;
  if (i < n) y[i] = (_Float16)fq_val(x[i], *amax);
}
__global__ void k_fq_ip(float* __restrict__ x, const float* __restrict__ amax, long long n) {
  long long i = blockIdx.x * 256LL + threadIdx.x;
  if (i < n) x[i] = fq_val(x[i], *amax);
}
__global__ void k_fq_add(float* __restrict__ h, const float* __restrict__ a,
                         const float* __restrict__ amax, long long n) {
  long long i = blockIdx.x * 256LL + threadIdx.x;
  if (i < n) h[i] += fq_val(a[i], *amax);
}

__global__ void k_patch(const float* __restrict__ x, const float* __restrict__ pw,
                        const float* __restrict__ pb, const float* __restrict__ am,
                        float* __restrict__ h) {
  long long idx = blockIdx.x * 256LL + threadIdx.x;
  if (idx >= (long long)M_ * D_) return;
  int d = (int)(idx % D_);
  long long r = idx / D_;
  int tok = (int)(r % SEQ_);
  int b = (int)(r / SEQ_);
  int c = tok / 50, p = tok % 50;
  float sx = am[0] * (1.f / 127.f) + 1e-8f;
  float sw = am[1] * (1.f / 127.f) + 1e-8f;
  const float* xs = x + ((size_t)b * C_ + c) * T_ + p * PATCH_;
  const float* ws = pw + d * PATCH_;
  float acc = 0.f;
  for (int i = 0; i < PATCH_; ++i) {
    float qx = fminf(fmaxf(rintf(xs[i] / sx), -128.f), 127.f) * sx;
    float qw = fminf(fmaxf(rintf(ws[i] / sw), -128.f), 127.f) * sw;
    acc += qx * qw;
  }
  h[idx] = acc + pb[d];
}

__global__ __launch_bounds__(256)
void k_ln_f16(const float* __restrict__ h, const float* __restrict__ g,
              const float* __restrict__ bta, _Float16* __restrict__ out, int rows) {
  int wave = threadIdx.x >> 5, lane = threadIdx.x & 31;
  int row = blockIdx.x * 8 + wave;
  if (row >= rows) return;
  const float* x = h + (size_t)row * D_;
  float v[6], s = 0.f, s2 = 0.f;
  for (int i = 0; i < 6; ++i) { v[i] = x[lane + 32 * i]; s += v[i]; s2 += v[i] * v[i]; }
  for (int off = 16; off; off >>= 1) { s += __shfl_xor(s, off); s2 += __shfl_xor(s2, off); }
  float mean = s * (1.f / D_);
  float var = s2 * (1.f / D_) - mean * mean;
  float inv = rsqrtf(var + 1e-5f);
  _Float16* o = out + (size_t)row * D_;
  for (int i = 0; i < 6; ++i) {
    int c = lane + 32 * i;
    o[c] = (_Float16)((v[i] - mean) * inv * g[c] + bta[c]);
  }
}
__global__ __launch_bounds__(256)
void k_ln_f32(const float* __restrict__ h, const float* __restrict__ g,
              const float* __restrict__ bta, float* __restrict__ out, int rows) {
  int wave = threadIdx.x >> 5, lane = threadIdx.x & 31;
  int row = blockIdx.x * 8 + wave;
  if (row >= rows) return;
  const float* x = h + (size_t)row * D_;
  float v[6], s = 0.f, s2 = 0.f;
  for (int i = 0; i < 6; ++i) { v[i] = x[lane + 32 * i]; s += v[i]; s2 += v[i] * v[i]; }
  for (int off = 16; off; off >>= 1) { s += __shfl_xor(s, off); s2 += __shfl_xor(s2, off); }
  float mean = s * (1.f / D_);
  float var = s2 * (1.f / D_) - mean * mean;
  float inv = rsqrtf(var + 1e-5f);
  float* o = out + (size_t)row * D_;
  for (int i = 0; i < 6; ++i) {
    int c = lane + 32 * i;
    o[c] = (v[i] - mean) * inv * g[c] + bta[c];
  }
}

__global__ __launch_bounds__(256)
void k_gemm_f16(const _Float16* __restrict__ A, const _Float16* __restrict__ W,
                const float* __restrict__ bias, float* __restrict__ C,
                int M, int N, int K, int act) {
  __shared__ __align__(16) _Float16 As[2][128 * 32];
  __shared__ __align__(16) _Float16 Ws[2][64 * 32];
  const int m0 = blockIdx.x * 128;
  const int n0 = blockIdx.y * 64;
  const int tid = threadIdx.x;
  const int wave = tid >> 5, lane = tid & 31;
  const int wm = (wave & 3) * 32;
  const int wn = (wave >> 2) * 32;
  const int hi = lane >> 4;
  const int ln16 = lane & 15;

  const int lr = tid >> 2;
  const int lc = (tid & 3) * 8;
  const _Float16* gA0 = A + (size_t)(m0 + lr) * K + lc;
  const _Float16* gA1 = A + (size_t)(m0 + 64 + lr) * K + lc;
  const _Float16* gW  = W + (size_t)(n0 + lr) * K + lc;
  const unsigned ldsA0 = (unsigned)(size_t)&As[0][lr * 32 + lc];
  const unsigned ldsA1 = (unsigned)(size_t)&As[0][(64 + lr) * 32 + lc];
  const unsigned ldsW  = (unsigned)(size_t)&Ws[0][lr * 32 + lc];
  constexpr unsigned ABUF = 128 * 32 * 2;
  constexpr unsigned WBUF = 64 * 32 * 2;

  v8f acc00 = {}, acc01 = {}, acc10 = {}, acc11 = {};
  const int nk = K / 32;

  async_cp16(ldsA0, gA0);
  async_cp16(ldsA1, gA1);
  async_cp16(ldsW,  gW);

  for (int i = 0; i < nk; ++i) {
    const int buf = i & 1;
    async_wait0();
    __syncthreads();
    if (i + 1 < nk) {
      const int k1 = (i + 1) * 32;
      const unsigned ab = (buf ^ 1) * ABUF, wb = (buf ^ 1) * WBUF;
      async_cp16(ldsA0 + ab, gA0 + k1);
      async_cp16(ldsA1 + ab, gA1 + k1);
      async_cp16(ldsW + wb,  gW + k1);
    }

    v16h a0, a1;
    {
      const _Float16* p0 = &As[buf][(wm + ln16) * 32 + hi * 8];
      ((uint4*)&a0)[0] = *(const uint4*)(p0);
      ((uint4*)&a0)[1] = *(const uint4*)(p0 + 16);
      const _Float16* p1 = &As[buf][(wm + 16 + ln16) * 32 + hi * 8];
      ((uint4*)&a1)[0] = *(const uint4*)(p1);
      ((uint4*)&a1)[1] = *(const uint4*)(p1 + 16);
    }
    v16h b0, b1;
    {
      const _Float16* q0 = &Ws[buf][(wn + ln16) * 32 + hi * 8];
      ((uint4*)&b0)[0] = *(const uint4*)(q0);
      ((uint4*)&b0)[1] = *(const uint4*)(q0 + 16);
      const _Float16* q1 = &Ws[buf][(wn + 16 + ln16) * 32 + hi * 8];
      ((uint4*)&b1)[0] = *(const uint4*)(q1);
      ((uint4*)&b1)[1] = *(const uint4*)(q1 + 16);
    }
    acc00 = __builtin_amdgcn_wmma_f32_16x16x32_f16(false, a0, false, b0, (short)0, acc00, false, false);
    acc01 = __builtin_amdgcn_wmma_f32_16x16x32_f16(false, a0, false, b1, (short)0, acc01, false, false);
    acc10 = __builtin_amdgcn_wmma_f32_16x16x32_f16(false, a1, false, b0, (short)0, acc10, false, false);
    acc11 = __builtin_amdgcn_wmma_f32_16x16x32_f16(false, a1, false, b1, (short)0, acc11, false, false);
  }

  const int rowb = m0 + wm + hi * 8;
  const int col = n0 + wn + ln16;
  const float bz0 = bias[col], bz1 = bias[col + 16];
  for (int r = 0; r < 8; ++r) {
    float v00 = acc00[r] + bz0;
    float v01 = acc01[r] + bz1;
    float v10 = acc10[r] + bz0;
    float v11 = acc11[r] + bz1;
    if (act == 1) { v00 = gelu_f(v00); v01 = gelu_f(v01); v10 = gelu_f(v10); v11 = gelu_f(v11); }
    C[(size_t)(rowb + r) * N + col]           = v00;
    C[(size_t)(rowb + r) * N + col + 16]      = v01;
    C[(size_t)(rowb + 16 + r) * N + col]      = v10;
    C[(size_t)(rowb + 16 + r) * N + col + 16] = v11;
  }
}

__global__ void k_quant_qkv(const float* __restrict__ qkv, const float* __restrict__ amax,
                            signed char* __restrict__ q8, signed char* __restrict__ k8,
                            signed char* __restrict__ vT8) {
  long long idx = blockIdx.x * 256LL + threadIdx.x;
  if (idx >= (long long)B_ * H_ * SPAD_ * DH_) return;
  int d = (int)(idx & 63);
  long long t = idx >> 6;
  int s = (int)(t % SPAD_); t /= SPAD_;
  int hh = (int)(t % H_);
  int b = (int)(t / H_);
  float sq = amax[0] * (1.f / 127.f) + 1e-8f;
  float sk = amax[1] * (1.f / 127.f) + 1e-8f;
  float sv = amax[2] * (1.f / 127.f) + 1e-8f;
  signed char qv = 0, kv = 0, vv = 0;
  if (s < SEQ_) {
    const float* p = qkv + ((size_t)(b * SEQ_ + s)) * (3 * D_) + hh * DH_ + d;
    qv = q_i8(p[0],       sq);
    kv = q_i8(p[D_],      sk);
    vv = q_i8(p[2 * D_],  sv);
  }
  size_t o = (((size_t)(b * H_ + hh)) * SPAD_ + s) * DH_ + d;
  q8[o] = qv;
  k8[o] = kv;
  vT8[(((size_t)(b * H_ + hh)) * DH_ + d) * SPAD_ + s] = vv;
}

__global__ __launch_bounds__(128)
void k_attn(const signed char* __restrict__ q8, const signed char* __restrict__ k8,
            const signed char* __restrict__ vT8, const float* __restrict__ amax,
            float* __restrict__ ctx) {
  __shared__ __align__(16) unsigned char pstage[4][16 * 64];
  const int wave = threadIdx.x >> 5, lane = threadIdx.x & 31;
  const int mt = blockIdx.x * 4 + wave;
  if (mt >= SEQ_ / 16) return;
  const int bh = blockIdx.y;
  const int b = bh / H_, h = bh % H_;
  const int m0 = mt * 16;
  const signed char* Q  = q8  + (size_t)bh * SPAD_ * DH_;
  const signed char* Kp = k8  + (size_t)bh * SPAD_ * DH_;
  const signed char* Vp = vT8 + (size_t)bh * DH_ * SPAD_;
  const float s_q = amax[0] * (1.f / 127.f) + 1e-8f;
  const float s_k = amax[1] * (1.f / 127.f) + 1e-8f;
  const float s_v = amax[2] * (1.f / 127.f) + 1e-8f;
  const float sscale = s_q * s_k * 0.125f;
  const int hi = lane >> 4, ln16 = lane & 15;

  v8i aq;
  {
    const signed char* qr = Q + (size_t)(m0 + ln16) * DH_ + hi * 16;
    ((int4*)&aq)[0] = *(const int4*)(qr);
    ((int4*)&aq)[1] = *(const int4*)(qr + 32);
  }

  float runm[8], Lrow[8];
  v8f cacc[4];
  for (int r = 0; r < 8; ++r) { runm[r] = -1e30f; Lrow[r] = 0.f; }
  { v8f z = {}; for (int d = 0; d < 4; ++d) cacc[d] = z; }
  unsigned char* ps = pstage[wave];

  for (int tc = 0; tc < SPAD_ / 64; ++tc) {
    const int nvalid = (tc < 6) ? 4 : 1;
    v8i sc_i[4];
    for (int tt = 0; tt < 4; ++tt) {
      if (tt < nvalid) {
        const int t0 = tc * 64 + tt * 16;
        v8i bk;
        const signed char* kr = Kp + (size_t)(t0 + ln16) * DH_ + hi * 16;
        ((int4*)&bk)[0] = *(const int4*)(kr);
        ((int4*)&bk)[1] = *(const int4*)(kr + 32);
        v8i zero = {};
        sc_i[tt] = __builtin_amdgcn_wmma_i32_16x16x64_iu8(true, aq, true, bk, zero, false, false);
      }
    }
    float cm[8], sf[4][8];
    for (int r = 0; r < 8; ++r) cm[r] = -1e30f;
    for (int tt = 0; tt < nvalid; ++tt)
      for (int r = 0; r < 8; ++r) {
        sf[tt][r] = (float)sc_i[tt][r] * sscale;
        cm[r] = fmaxf(cm[r], sf[tt][r]);
      }
    for (int off = 1; off < 16; off <<= 1)
      for (int r = 0; r < 8; ++r) cm[r] = fmaxf(cm[r], __shfl_xor(cm[r], off));
    float lsum[8];
    for (int r = 0; r < 8; ++r) {
      float nm = fmaxf(runm[r], cm[r]);
      float resc = __expf(runm[r] - nm);
      runm[r] = nm;
      Lrow[r] *= resc;
      for (int d = 0; d < 4; ++d) cacc[d][r] *= resc;
      lsum[r] = 0.f;
    }
    for (int tt = 0; tt < 4; ++tt)
      for (int r = 0; r < 8; ++r) {
        unsigned char pu = 0;
        if (tt < nvalid) {
          float p = __expf(sf[tt][r] - runm[r]);
          lsum[r] += p;
          pu = (unsigned char)fminf(fmaxf(rintf(p * 255.f), 0.f), 255.f);
        }
        ps[(r + hi * 8) * 64 + tt * 16 + ln16] = pu;
      }
    for (int off = 1; off < 16; off <<= 1)
      for (int r = 0; r < 8; ++r) lsum[r] += __shfl_xor(lsum[r], off);
    for (int r = 0; r < 8; ++r) Lrow[r] += lsum[r];
    asm volatile("s_wait_dscnt 0" ::: "memory");

    v8i ap;
    {
      const unsigned char* pr = ps + ln16 * 64 + hi * 16;
      ((int4*)&ap)[0] = *(const int4*)(pr);
      ((int4*)&ap)[1] = *(const int4*)(pr + 32);
    }
    for (int d = 0; d < 4; ++d) {
      v8i bv;
      const signed char* vr = Vp + (size_t)(d * 16 + ln16) * SPAD_ + tc * 64 + hi * 16;
      ((int4*)&bv)[0] = *(const int4*)(vr);
      ((int4*)&bv)[1] = *(const int4*)(vr + 32);
      v8i zero = {};
      v8i ci = __builtin_amdgcn_wmma_i32_16x16x64_iu8(false, ap, true, bv, zero, false, false);
      for (int r = 0; r < 8; ++r) cacc[d][r] += (float)ci[r];
    }
  }

  for (int d = 0; d < 4; ++d)
    for (int r = 0; r < 8; ++r) {
      int row = m0 + r + hi * 8;
      int col = h * DH_ + d * 16 + ln16;
      ctx[((size_t)b * SEQ_ + row) * D_ + col] = cacc[d][r] * s_v / (255.f * Lrow[r]);
    }
}

__global__ void k_pool(const float* __restrict__ hn, float* __restrict__ pooled) {
  int idx = blockIdx.x * 256 + threadIdx.x;
  if (idx >= B_ * D_) return;
  int b = idx / D_, d = idx % D_;
  const float* p = hn + (size_t)b * SEQ_ * D_ + d;
  float s = 0.f;
  for (int t = 0; t < SEQ_; ++t) s += p[(size_t)t * D_];
  pooled[idx] = s * (1.f / SEQ_);
}

__global__ void k_clf(const float* __restrict__ pooled, const float* __restrict__ cw,
                      const float* __restrict__ amax_w, const float* __restrict__ cb,
                      float* __restrict__ out) {
  int idx = blockIdx.x * 64 + threadIdx.x;
  if (idx >= B_ * NCLS_) return;
  int b = idx / NCLS_, c = idx % NCLS_;
  float sw = amax_w[0] * (1.f / 127.f) + 1e-8f;
  const float* p = pooled + (size_t)b * D_;
  const float* w = cw + (size_t)c * D_;
  float acc = 0.f;
  for (int d = 0; d < D_; ++d) {
    float qw = fminf(fmaxf(rintf(w[d] / sw), -128.f), 127.f) * sw;
    acc += p[d] * qw;
  }
  out[idx] = acc + cb[c];
}

static inline size_t alignup(size_t x) { return (x + 255) & ~(size_t)255; }

extern "C" void kernel_launch(void* const* d_in, const int* in_sizes, int n_in,
                              void* d_out, int out_size, void* d_ws, size_t ws_size,
                              hipStream_t stream) {
  (void)in_sizes; (void)n_in; (void)out_size; (void)ws_size;
  const float* x_in    = (const float*)d_in[0];
  const float* patch_w = (const float*)d_in[1];
  const float* patch_b = (const float*)d_in[2];
  const float* ln1_g   = (const float*)d_in[3];
  const float* ln1_b   = (const float*)d_in[4];
  const float* w_qkv   = (const float*)d_in[5];
  const float* b_qkv   = (const float*)d_in[6];
  const float* w_o     = (const float*)d_in[7];
  const float* b_o     = (const float*)d_in[8];
  const float* ln2_g   = (const float*)d_in[9];
  const float* ln2_b   = (const float*)d_in[10];
  const float* w1      = (const float*)d_in[11];
  const float* b1      = (const float*)d_in[12];
  const float* w2      = (const float*)d_in[13];
  const float* b2      = (const float*)d_in[14];
  const float* lnf_g   = (const float*)d_in[15];
  const float* lnf_b   = (const float*)d_in[16];
  const float* clf_w   = (const float*)d_in[17];
  const float* clf_b   = (const float*)d_in[18];
  float* out = (float*)d_out;

  char* ws = (char*)d_ws;
  size_t off = 0;
  auto take = [&](size_t bytes) -> char* { char* p = ws + off; off = alignup(off + bytes); return p; };
  float*    scales = (float*)take(64 * sizeof(float));
  float*    h      = (float*)take((size_t)M_ * D_ * 4);
  _Float16* xn16   = (_Float16*)take((size_t)M_ * D_ * 2);
  float*    big1   = (float*)take((size_t)M_ * HID_ * 4);
  char*     dreg   = take((size_t)M_ * D_ * 4);
  float*    big2   = (float*)take((size_t)M_ * HID_ * 2);
  _Float16* wq16   = (_Float16*)take((size_t)L_ * 3 * D_ * D_ * 2);
  _Float16* wo16   = (_Float16*)take((size_t)L_ * D_ * D_ * 2);
  _Float16* w116   = (_Float16*)take((size_t)L_ * HID_ * D_ * 2);
  _Float16* w216   = (_Float16*)take((size_t)L_ * D_ * HID_ * 2);
  float*    pooled = (float*)take((size_t)B_ * D_ * 4);

  const size_t QSZ = (size_t)B_ * H_ * SPAD_ * DH_;
  signed char* q8  = (signed char*)dreg;
  signed char* k8  = q8 + QSZ;
  signed char* vT8 = k8 + QSZ;
  _Float16* ctx16  = (_Float16*)dreg;
  float*    m2     = (float*)dreg;
  float*    hn     = (float*)dreg;
  float* ctxf = big2;
  float* attn = big2;
  _Float16* m16 = (_Float16*)big2;

  auto absmax = [&](const float* p, long long n, float* slot) {
    k_absmax<<<1024, 256, 0, stream>>>(p, n, slot);
  };
  auto fq16 = [&](const float* p, const float* slot, _Float16* y, long long n) {
    k_fq16<<<(unsigned)((n + 255) / 256), 256, 0, stream>>>(p, slot, y, n);
  };

  k_zeroN<<<1, 64, 0, stream>>>(scales, 64);

  absmax(x_in,    (long long)B_ * C_ * T_,  scales + 0);
  absmax(patch_w, (long long)D_ * PATCH_,   scales + 1);
  absmax(clf_w,   (long long)NCLS_ * D_,    scales + 42);
  for (int i = 0; i < L_; ++i) {
    const long long nq = (long long)3 * D_ * D_, no = (long long)D_ * D_;
    const long long n1 = (long long)HID_ * D_,   n2 = (long long)D_ * HID_;
    absmax(w_qkv + (size_t)i * nq, nq, scales + 10 + i);
    absmax(w_o   + (size_t)i * no, no, scales + 18 + i);
    absmax(w1    + (size_t)i * n1, n1, scales + 26 + i);
    absmax(w2    + (size_t)i * n2, n2, scales + 34 + i);
    fq16(w_qkv + (size_t)i * nq, scales + 10 + i, wq16 + (size_t)i * nq, nq);
    fq16(w_o   + (size_t)i * no, scales + 18 + i, wo16 + (size_t)i * no, no);
    fq16(w1    + (size_t)i * n1, scales + 26 + i, w116 + (size_t)i * n1, n1);
    fq16(w2    + (size_t)i * n2, scales + 34 + i, w216 + (size_t)i * n2, n2);
  }

  const long long nh = (long long)M_ * D_;
  k_patch<<<(unsigned)((nh + 255) / 256), 256, 0, stream>>>(x_in, patch_w, patch_b, scales, h);
  absmax(h, nh, scales + 2);
  k_fq_ip<<<(unsigned)((nh + 255) / 256), 256, 0, stream>>>(h, scales + 2, nh);

  for (int i = 0; i < L_; ++i) {
    k_zeroN<<<1, 32, 0, stream>>>(scales + 3, 7);
    k_ln_f16<<<M_ / 8, 256, 0, stream>>>(h, ln1_g + i * D_, ln1_b + i * D_, xn16, M_);
    k_gemm_f16<<<dim3(M_ / 128, (3 * D_) / 64), 256, 0, stream>>>(
        xn16, wq16 + (size_t)i * 3 * D_ * D_, b_qkv + (size_t)i * 3 * D_, big1, M_, 3 * D_, D_, 0);
    k_absmax3<<<1024, 256, 0, stream>>>(big1, (long long)M_, scales + 3);
    k_quant_qkv<<<(unsigned)(QSZ / 256), 256, 0, stream>>>(big1, scales + 3, q8, k8, vT8);
    k_attn<<<dim3(7, B_ * H_), 128, 0, stream>>>(q8, k8, vT8, scales + 3, ctxf);
    absmax(ctxf, nh, scales + 6);
    fq16(ctxf, scales + 6, ctx16, nh);
    k_gemm_f16<<<dim3(M_ / 128, D_ / 64), 256, 0, stream>>>(
        ctx16, wo16 + (size_t)i * D_ * D_, b_o + (size_t)i * D_, attn, M_, D_, D_, 0);
    absmax(attn, nh, scales + 7);
    k_fq_add<<<(unsigned)((nh + 255) / 256), 256, 0, stream>>>(h, attn, scales + 7, nh);
    const long long nm = (long long)M_ * HID_;
    k_ln_f16<<<M_ / 8, 256, 0, stream>>>(h, ln2_g + i * D_, ln2_b + i * D_, xn16, M_);
    k_gemm_f16<<<dim3(M_ / 128, HID_ / 64), 256, 0, stream>>>(
        xn16, w116 + (size_t)i * HID_ * D_, b1 + (size_t)i * HID_, big1, M_, HID_, D_, 1);
    absmax(big1, nm, scales + 8);
    fq16(big1, scales + 8, m16, nm);
    k_gemm_f16<<<dim3(M_ / 128, D_ / 64), 256, 0, stream>>>(
        m16, w216 + (size_t)i * D_ * HID_, b2 + (size_t)i * D_, m2, M_, D_, HID_, 0);
    absmax(m2, nh, scales + 9);
    k_fq_add<<<(unsigned)((nh + 255) / 256), 256, 0, stream>>>(h, m2, scales + 9, nh);
  }

  k_ln_f32<<<M_ / 8, 256, 0, stream>>>(h, lnf_g, lnf_b, hn, M_);
  k_pool<<<(B_ * D_ + 255) / 256, 256, 0, stream>>>(hn, pooled);
  absmax(pooled, (long long)B_ * D_, scales + 43);
  k_fq_ip<<<(B_ * D_ + 255) / 256, 256, 0, stream>>>(pooled, scales + 43, (long long)B_ * D_);
  k_clf<<<(B_ * NCLS_ + 63) / 64, 64, 0, stream>>>(pooled, clf_w, scales + 42, clf_b, out);
}
